// GraphBlock_20194936225828
// MI455X (gfx1250) — matrix operand order repaired and checked by reading, not compiled, not run on hardware
//
#include <hip/hip_runtime.h>
#include <hip/hip_bf16.h>
#include <math.h>

typedef __bf16 bf16;
typedef __bf16 v8bf  __attribute__((ext_vector_type(8)));
typedef __bf16 v16bf __attribute__((ext_vector_type(16)));
typedef float  v8f   __attribute__((ext_vector_type(8)));
typedef int    v4i   __attribute__((vector_size(4 * sizeof(int))));

typedef __attribute__((address_space(1))) v4i*  as1_v4i;
typedef __attribute__((address_space(3))) v4i*  as3_v4i;
typedef __attribute__((address_space(1))) const bf16* as1_cbf;
typedef __attribute__((address_space(3))) bf16* as3_bf;

#define WMMA_BF16(a, b, c) \
  __builtin_amdgcn_wmma_f32_16x16x32_bf16(false, (a), false, (b), (short)0, (c), false, false)

#if defined(__has_builtin)
#if __has_builtin(__builtin_amdgcn_global_load_async_to_lds_b128) && \
    __has_builtin(__builtin_amdgcn_s_wait_asynccnt)
#define HAS_ASYNC 1
#endif
#endif
#ifndef HAS_ASYNC
#define HAS_ASYNC 0
#endif

#define B_   4
#define S_   2048
#define D_   1024
#define NH_  16
#define HD_  64
#define FF_  4096
#define E_   4096

__device__ __forceinline__ v16bf mk16(v8bf lo, v8bf hi) {
  return __builtin_shufflevector(lo, hi, 0, 1, 2, 3, 4, 5, 6, 7,
                                 8, 9, 10, 11, 12, 13, 14, 15);
}

// ---------------- elementwise / setup kernels ----------------

__global__ void zero_f32(float* __restrict__ p, size_t n) {
  size_t i = (size_t)blockIdx.x * blockDim.x + threadIdx.x;
  size_t stride = (size_t)gridDim.x * blockDim.x;
  for (; i < n; i += stride) p[i] = 0.f;
}

__global__ void scatter_bias(const int* __restrict__ edges, float* __restrict__ bias) {
  int tid = blockIdx.x * blockDim.x + threadIdx.x;
  if (tid >= B_ * E_) return;
  int b = tid / E_, e = tid % E_;
  int src = edges[((size_t)b * E_ + e) * 2 + 0];
  int dst = edges[((size_t)b * E_ + e) * 2 + 1];
  atomicAdd(&bias[((size_t)b * S_ + src) * S_ + dst], 1.0f);
}

// f32 W[K,N]  ->  bf16 Wt[N,K]  (LDS tile transpose, coalesced both sides)
__global__ void cvt_transpose_bf16(const float* __restrict__ W, bf16* __restrict__ Wt,
                                   int K, int N) {
  __shared__ float tile[32][33];
  int k0 = blockIdx.x * 32, n0 = blockIdx.y * 32;
  int tx = threadIdx.x & 31, ty = threadIdx.x >> 5;  // 32 x 8
#pragma unroll
  for (int i = 0; i < 4; i++)
    tile[ty + i * 8][tx] = W[(size_t)(k0 + ty + i * 8) * N + n0 + tx];
  __syncthreads();
#pragma unroll
  for (int i = 0; i < 4; i++)
    Wt[(size_t)(n0 + ty + i * 8) * K + k0 + tx] = (bf16)tile[tx][ty + i * 8];
}

// bf16 V[B*S, D] -> bf16 Vt[B*D, S]
__global__ void transpose_v(const bf16* __restrict__ V, bf16* __restrict__ Vt) {
  __shared__ bf16 tile[32][33];
  int s0 = blockIdx.x * 32, d0 = blockIdx.y * 32, b = blockIdx.z;
  int tx = threadIdx.x & 31, ty = threadIdx.x >> 5;
#pragma unroll
  for (int i = 0; i < 4; i++)
    tile[ty + i * 8][tx] = V[((size_t)b * S_ + s0 + ty + i * 8) * D_ + d0 + tx];
  __syncthreads();
#pragma unroll
  for (int i = 0; i < 4; i++)
    Vt[((size_t)b * D_ + d0 + ty + i * 8) * S_ + s0 + tx] = tile[tx][ty + i * 8];
}

// ---------------- layernorm: f32 row -> bf16 row ----------------

__global__ void ln_kernel(const float* __restrict__ x, const float* __restrict__ g,
                          const float* __restrict__ beta, bf16* __restrict__ out) {
  __shared__ float sh1[256], sh2[256];
  int row = blockIdx.x, t = threadIdx.x;
  const float* xr = x + (size_t)row * D_;
  float v0 = xr[t], v1 = xr[t + 256], v2 = xr[t + 512], v3 = xr[t + 768];
  sh1[t] = v0 + v1 + v2 + v3;
  sh2[t] = v0 * v0 + v1 * v1 + v2 * v2 + v3 * v3;
  __syncthreads();
  for (int o = 128; o > 0; o >>= 1) {
    if (t < o) { sh1[t] += sh1[t + o]; sh2[t] += sh2[t + o]; }
    __syncthreads();
  }
  float mu = sh1[0] * (1.f / D_);
  float var = sh2[0] * (1.f / D_) - mu * mu;
  float rs = rsqrtf(var + 1e-5f);
  bf16* orow = out + (size_t)row * D_;
  orow[t]       = (bf16)((v0 - mu) * rs * g[t]       + beta[t]);
  orow[t + 256] = (bf16)((v1 - mu) * rs * g[t + 256] + beta[t + 256]);
  orow[t + 512] = (bf16)((v2 - mu) * rs * g[t + 512] + beta[t + 512]);
  orow[t + 768] = (bf16)((v3 - mu) * rs * g[t + 768] + beta[t + 768]);
}

// ---------------- WMMA GEMM: C[M,N] = A[M,K] @ Wt[N,K]^T + epilogue ----------------
// 4 waves / block, 128x64 C tile (each wave 32x64 -> 8 WMMA per k-step).
// Wt k-slab double-buffered in LDS (stride 40 bf16 = 80B, 16B-aligned,
// conflict-free); fill of next slab issued before this slab's WMMAs
// (global_load_async_to_lds_b128 when available, reg-bounce otherwise);
// one barrier per k-step.
// EPI 0: bf16(c)   1: f32 resid+c   2: bf16 gelu(c+bias)   3: f32 resid+c+bias

template <int EPI>
__global__ void gemm_wmma(const bf16* __restrict__ A, const bf16* __restrict__ Wt,
                          const float* __restrict__ bias, const float* __restrict__ resid,
                          float* __restrict__ outF, bf16* __restrict__ outB,
                          int M, int N, int K) {
  constexpr int SLAB = 64 * 40;
  __shared__ alignas(16) bf16 smem[2 * SLAB];
  int tid = threadIdx.x;
  int lane = tid & 31, wave = tid >> 5;
  int hl = lane >> 4, ln = lane & 15;
  int m0 = blockIdx.x * 128 + wave * 32;
  int n0 = blockIdx.y * 64;

  v8f acc[2][4];
#pragma unroll
  for (int w = 0; w < 2; w++)
#pragma unroll
    for (int t = 0; t < 4; t++)
#pragma unroll
      for (int i = 0; i < 8; i++) acc[w][t][i] = 0.f;

  const bf16* aRow0 = A + (size_t)(m0 + ln) * K;
  const bf16* aRow1 = A + (size_t)(m0 + 16 + ln) * K;
  int wn = tid >> 1, wc = tid & 1;  // 64 n-rows x 2 chunks of 16 bf16
  const bf16* wSrc = Wt + (size_t)(n0 + wn) * K + wc * 16;
  int ldsOff = wn * 40 + wc * 16;

  auto fill = [&](int k0, int buf) {
#if HAS_ASYNC
    as1_cbf g0 = (as1_cbf)(wSrc + k0);
    as3_bf  l0 = (as3_bf)(smem + buf * SLAB + ldsOff);
    __builtin_amdgcn_global_load_async_to_lds_b128((as1_v4i)g0, (as3_v4i)l0, 0, 0);
    __builtin_amdgcn_global_load_async_to_lds_b128((as1_v4i)(g0 + 8), (as3_v4i)(l0 + 8), 0, 0);
#else
    v8bf w0 = *(const v8bf*)(wSrc + k0);
    v8bf w1v = *(const v8bf*)(wSrc + k0 + 8);
    *(v8bf*)(smem + buf * SLAB + ldsOff) = w0;
    *(v8bf*)(smem + buf * SLAB + ldsOff + 8) = w1v;
#endif
  };

  fill(0, 0);
#if HAS_ASYNC
  __builtin_amdgcn_s_wait_asynccnt(0);
#endif
  __syncthreads();

  int cur = 0;
  for (int k0 = 0; k0 < K; k0 += 32) {
    bool nxt = (k0 + 32) < K;
    if (nxt) fill(k0 + 32, 1 - cur);  // overlaps with WMMAs below

    const bf16* base = &smem[cur * SLAB];
    v16bf a0 = mk16(*(const v8bf*)(aRow0 + k0 + hl * 8),
                    *(const v8bf*)(aRow0 + k0 + 16 + hl * 8));
    v16bf a1 = mk16(*(const v8bf*)(aRow1 + k0 + hl * 8),
                    *(const v8bf*)(aRow1 + k0 + 16 + hl * 8));
#pragma unroll
    for (int t = 0; t < 4; t++) {
      const bf16* bp = base + (t * 16 + ln) * 40 + hl * 8;
      v16bf bfv = mk16(*(const v8bf*)bp, *(const v8bf*)(bp + 16));
      acc[0][t] = WMMA_BF16(a0, bfv, acc[0][t]);
      acc[1][t] = WMMA_BF16(a1, bfv, acc[1][t]);
    }

#if HAS_ASYNC
    if (nxt) __builtin_amdgcn_s_wait_asynccnt(0);
#endif
    __syncthreads();  // next slab visible; everyone done reading cur
    cur ^= 1;
  }

#pragma unroll
  for (int w = 0; w < 2; w++) {
#pragma unroll
    for (int t = 0; t < 4; t++) {
#pragma unroll
      for (int r = 0; r < 8; r++) {
        int row = m0 + w * 16 + r + hl * 8;
        int col = n0 + t * 16 + ln;
        size_t idx = (size_t)row * N + col;
        float c = acc[w][t][r];
        if (EPI == 0) {
          outB[idx] = (bf16)c;
        } else if (EPI == 1) {
          outF[idx] = resid[idx] + c;
        } else if (EPI == 2) {
          float vv = c + bias[col];
          outB[idx] = (bf16)(0.5f * vv * (1.f + erff(vv * 0.70710678118f)));
        } else {
          outF[idx] = resid[idx] + c + bias[col];
        }
      }
    }
  }
}

// ---------------- flash attention with graph bias ----------------
// grid = (S/16, NH, B); one wave per block, 16 Q rows x one head.
// Keys streamed 32 at a time; all operand fragments are contiguous b128 loads
// (V pre-transposed to [B*D, S]); 1/sqrt(hd) folded into f32 scores.

__global__ void attn_kernel(const bf16* __restrict__ Q, const bf16* __restrict__ Km,
                            const bf16* __restrict__ Vt, const float* __restrict__ bias,
                            const int* __restrict__ amask, bf16* __restrict__ O) {
  __shared__ alignas(16) bf16 pbuf[16 * 40];  // padded stride: 80B rows
  int lane = threadIdx.x, hl = lane >> 4, ln = lane & 15;
  int q0 = blockIdx.x * 16, h = blockIdx.y, b = blockIdx.z;
  size_t baseBS = (size_t)b * S_;
  size_t hoff = (size_t)h * HD_;

  const bf16* qrow = Q + (baseBS + q0 + ln) * D_ + hoff;
  v16bf aq0 = mk16(*(const v8bf*)(qrow + hl * 8), *(const v8bf*)(qrow + 16 + hl * 8));
  v16bf aq1 = mk16(*(const v8bf*)(qrow + 32 + hl * 8), *(const v8bf*)(qrow + 48 + hl * 8));

  v8f acc[4];
#pragma unroll
  for (int t = 0; t < 4; t++)
#pragma unroll
    for (int i = 0; i < 8; i++) acc[t][i] = 0.f;

  float mrun[8], lrun[8];
#pragma unroll
  for (int r = 0; r < 8; r++) { mrun[r] = -1e30f; lrun[r] = 0.f; }

  const float* biasB = bias + (size_t)b * S_ * S_;
  const int* amB = amask + baseBS;
  int jEnd = q0 + 16;  // causal: last needed key is q0+15

  for (int jj = 0; jj < jEnd; jj += 32) {
    v8f s0, s1;
#pragma unroll
    for (int i = 0; i < 8; i++) { s0[i] = 0.f; s1[i] = 0.f; }

    {  // scores: Q @ K^T for keys [jj, jj+16) and [jj+16, jj+32)
      const bf16* k0r = Km + (baseBS + jj + ln) * D_ + hoff;
      const bf16* k1r = Km + (baseBS + jj + 16 + ln) * D_ + hoff;
      v16bf bk;
      bk = mk16(*(const v8bf*)(k0r + hl * 8), *(const v8bf*)(k0r + 16 + hl * 8));
      s0 = WMMA_BF16(aq0, bk, s0);
      bk = mk16(*(const v8bf*)(k0r + 32 + hl * 8), *(const v8bf*)(k0r + 48 + hl * 8));
      s0 = WMMA_BF16(aq1, bk, s0);
      bk = mk16(*(const v8bf*)(k1r + hl * 8), *(const v8bf*)(k1r + 16 + hl * 8));
      s1 = WMMA_BF16(aq0, bk, s1);
      bk = mk16(*(const v8bf*)(k1r + 32 + hl * 8), *(const v8bf*)(k1r + 48 + hl * 8));
      s1 = WMMA_BF16(aq1, bk, s1);
    }

    // bias + masks + online softmax (row = r + hl*8, cols across 16-lane halves)
#pragma unroll
    for (int r = 0; r < 8; r++) {
      int rowq = q0 + r + hl * 8;
      int kg0 = jj + ln, kg1 = jj + 16 + ln;
      float sv0 = -1e30f, sv1 = -1e30f;
      if (kg0 <= rowq && amB[kg0] != 0)
        sv0 = s0[r] * 0.125f + fminf(fmaxf(biasB[(size_t)rowq * S_ + kg0], -5.f), 5.f);
      if (kg1 <= rowq && amB[kg1] != 0)
        sv1 = s1[r] * 0.125f + fminf(fmaxf(biasB[(size_t)rowq * S_ + kg1], -5.f), 5.f);

      float rmax = fmaxf(sv0, sv1);
#pragma unroll
      for (int off = 8; off > 0; off >>= 1) rmax = fmaxf(rmax, __shfl_xor(rmax, off, 16));
      float mnew = fmaxf(mrun[r], rmax);
      float alpha = __expf(mrun[r] - mnew);
      float p0 = (sv0 > -1e29f) ? __expf(sv0 - mnew) : 0.f;
      float p1 = (sv1 > -1e29f) ? __expf(sv1 - mnew) : 0.f;
      float rsum = p0 + p1;
#pragma unroll
      for (int off = 8; off > 0; off >>= 1) rsum += __shfl_xor(rsum, off, 16);
      lrun[r] = lrun[r] * alpha + rsum;
      mrun[r] = mnew;
#pragma unroll
      for (int t = 0; t < 4; t++) acc[t][r] *= alpha;
      pbuf[(r + hl * 8) * 40 + ln] = (bf16)p0;
      pbuf[(r + hl * 8) * 40 + 16 + ln] = (bf16)p1;
    }
    __syncthreads();

    // P tile (C-layout -> A-layout bounce), then P @ V
    v16bf pa = mk16(*(const v8bf*)&pbuf[ln * 40 + hl * 8],
                    *(const v8bf*)&pbuf[ln * 40 + 16 + hl * 8]);
    __syncthreads();

#pragma unroll
    for (int t = 0; t < 4; t++) {
      const bf16* vrow = Vt + ((size_t)b * D_ + hoff + t * 16 + ln) * S_ + jj + hl * 8;
      v16bf bv = mk16(*(const v8bf*)vrow, *(const v8bf*)(vrow + 16));
      acc[t] = WMMA_BF16(pa, bv, acc[t]);
    }
  }

#pragma unroll
  for (int t = 0; t < 4; t++)
#pragma unroll
    for (int r = 0; r < 8; r++) {
      float o = acc[t][r] / fmaxf(lrun[r], 1e-20f);
      O[(baseBS + q0 + r + hl * 8) * D_ + hoff + t * 16 + ln] = (bf16)o;
    }
}

// ---------------- launch ----------------

extern "C" void kernel_launch(void* const* d_in, const int* in_sizes, int n_in,
                              void* d_out, int out_size, void* d_ws, size_t ws_size,
                              hipStream_t stream) {
  (void)in_sizes; (void)n_in; (void)out_size; (void)ws_size;
  const float* x    = (const float*)d_in[0];
  const float* wq   = (const float*)d_in[1];
  const float* wk   = (const float*)d_in[2];
  const float* wv   = (const float*)d_in[3];
  const float* wo   = (const float*)d_in[4];
  const float* lg1  = (const float*)d_in[5];
  const float* lb1  = (const float*)d_in[6];
  const float* lg2  = (const float*)d_in[7];
  const float* lb2  = (const float*)d_in[8];
  const float* w1   = (const float*)d_in[9];
  const float* b1   = (const float*)d_in[10];
  const float* w2   = (const float*)d_in[11];
  const float* b2   = (const float*)d_in[12];
  const int* amask  = (const int*)d_in[13];
  const int* edges  = (const int*)d_in[14];
  float* out = (float*)d_out;

  char* ws = (char*)d_ws;
  const size_t MB = 1ull << 20;
  bf16* h1   = (bf16*)(ws);              // 16 MB  LN1 out
  bf16* wqb  = (bf16*)(ws + 16 * MB);    //  2 MB  (transposed)
  bf16* wkb  = (bf16*)(ws + 18 * MB);
  bf16* wvb  = (bf16*)(ws + 20 * MB);
  bf16* wob  = (bf16*)(ws + 22 * MB);
  bf16* w1b  = (bf16*)(ws + 24 * MB);    //  8 MB  (transposed)
  bf16* w2b  = (bf16*)(ws + 32 * MB);    //  8 MB  (transposed)
  bf16* qb   = (bf16*)(ws + 40 * MB);    // 16 MB
  bf16* kb   = (bf16*)(ws + 56 * MB);
  bf16* vtb  = (bf16*)(ws + 72 * MB);    // 16 MB  V transposed [B*D, S]
  bf16* aob  = (bf16*)(ws + 88 * MB);    // 16 MB  attention out
  float* x1  = (float*)(ws + 104 * MB);  // 32 MB
  bf16* h2   = (bf16*)(ws + 136 * MB);   // 16 MB  LN2 out
  bf16* mid  = (bf16*)(ws + 152 * MB);   // 64 MB  MLP hidden (also temp for raw V)
  float* bias = (float*)(ws + 216 * MB); // 64 MB  [B,S,S]

  const int rows = B_ * S_;  // 8192

  // weight convert + transpose: f32 W[K,N] -> bf16 Wt[N,K]
  dim3 tDD(D_ / 32, D_ / 32);
  cvt_transpose_bf16<<<tDD, 256, 0, stream>>>(wq, wqb, D_, D_);
  cvt_transpose_bf16<<<tDD, 256, 0, stream>>>(wk, wkb, D_, D_);
  cvt_transpose_bf16<<<tDD, 256, 0, stream>>>(wv, wvb, D_, D_);
  cvt_transpose_bf16<<<tDD, 256, 0, stream>>>(wo, wob, D_, D_);
  cvt_transpose_bf16<<<dim3(D_ / 32, FF_ / 32), 256, 0, stream>>>(w1, w1b, D_, FF_);
  cvt_transpose_bf16<<<dim3(FF_ / 32, D_ / 32), 256, 0, stream>>>(w2, w2b, FF_, D_);

  // LN1
  ln_kernel<<<rows, 256, 0, stream>>>(x, lg1, lb1, h1);

  // Q, K, V projections (V lands in temp, then transposed to [B*D, S])
  dim3 gDD(rows / 128, D_ / 64);
  gemm_wmma<0><<<gDD, 128, 0, stream>>>(h1, wqb, nullptr, nullptr, nullptr, qb, rows, D_, D_);
  gemm_wmma<0><<<gDD, 128, 0, stream>>>(h1, wkb, nullptr, nullptr, nullptr, kb, rows, D_, D_);
  gemm_wmma<0><<<gDD, 128, 0, stream>>>(h1, wvb, nullptr, nullptr, nullptr, (bf16*)mid, rows, D_, D_);
  transpose_v<<<dim3(S_ / 32, D_ / 32, B_), 256, 0, stream>>>((bf16*)mid, vtb);

  // graph bias scatter (zeroed every call -> deterministic)
  zero_f32<<<4096, 256, 0, stream>>>(bias, (size_t)B_ * S_ * S_);
  scatter_bias<<<(B_ * E_ + 255) / 256, 256, 0, stream>>>(edges, bias);

  // attention
  dim3 gA(S_ / 16, NH_, B_);
  attn_kernel<<<gA, 32, 0, stream>>>(qb, kb, vtb, bias, amask, aob);

  // x1 = x + attn_out @ wo
  gemm_wmma<1><<<gDD, 128, 0, stream>>>(aob, wob, nullptr, x, x1, nullptr, rows, D_, D_);

  // LN2
  ln_kernel<<<rows, 256, 0, stream>>>(x1, lg2, lb2, h2);

  // MLP
  dim3 gDH(rows / 128, FF_ / 64);
  gemm_wmma<2><<<gDH, 128, 0, stream>>>(h2, w1b, b1, nullptr, nullptr, mid, rows, FF_, D_);
  gemm_wmma<3><<<gDD, 128, 0, stream>>>(mid, w2b, b2, x1, out, nullptr, rows, D_, FF_);
}
